// AdaptiveEmbedding_71150428225573
// MI455X (gfx1250) — matrix operand order repaired and checked by reading, not compiled, not run on hardware
//
#include <hip/hip_runtime.h>

// ---------------------------------------------------------------------------
// AdaptiveEmbedding for MI455X (gfx1250, wave32, WMMA + TDM).
//
//   out[n,:] = w_c @ emb_c[idx[n]-lo_c],  c0:[0,20000) h=1024,
//   c1:[20000,40000) h=256, c2:[40000,50257) h=64
//
// Memory floor ~5us (110MB @ 23.3TB/s). Design:
//   1) partition tokens by cluster (atomics) -> dense same-cluster tiles
//   2) bf16x3 split-GEMM (hi*hi + hi*lo + lo*hi) on v_wmma_f32_16x16x32_bf16,
//      f32 accumulate -> ~f32 accuracy at bf16 matrix rate
//   3) A-rows gathered by the Tensor Data Mover in gather mode (16 row ids
//      per descriptor) straight into LDS, double-buffered, TENSORcnt-tracked
// Fallback (tiny ws): padded f32 16x16x4 WMMA kernel.
// ---------------------------------------------------------------------------

typedef __bf16 bf16_t;
typedef __bf16 v16bf __attribute__((ext_vector_type(16)));
typedef __bf16 bfx8  __attribute__((ext_vector_type(8)));
typedef __bf16 bfx4  __attribute__((ext_vector_type(4)));
typedef float  v8f   __attribute__((ext_vector_type(8)));
typedef float  v2f   __attribute__((ext_vector_type(2)));
typedef unsigned int u32x4 __attribute__((ext_vector_type(4)));
typedef int    i32x4 __attribute__((ext_vector_type(4)));
typedef int    i32x8 __attribute__((ext_vector_type(8)));

#if __has_builtin(__builtin_amdgcn_tensor_load_to_lds)
#define USE_TDM 1
#else
#define USE_TDM 0
#endif

#define D_MODEL 1024

constexpr int cLO[3]  = {0,     20000, 40000};
constexpr int cHIe[3] = {20000, 40000, 50257};
constexpr int cH[3]   = {1024,  256,   64};
constexpr int cVS[3]  = {20000, 20000, 10257};   // rows per embedding table

// ======================== helper device functions ==========================

__device__ __forceinline__ v16bf cat8(bfx8 a, bfx8 b) {
    v16bf r;
#pragma unroll
    for (int i = 0; i < 8; ++i) { r[i] = a[i]; r[8 + i] = b[i]; }
    return r;
}

__device__ __forceinline__ v16bf ld16bf(const bf16_t* __restrict__ p) {
    return cat8(*(const bfx8*)p, *(const bfx8*)(p + 8));
}

// split 16 consecutive f32 into hi/lo bf16 fragments
__device__ __forceinline__ void cvt16(const float* __restrict__ p,
                                      v16bf& hi, v16bf& lo) {
#pragma unroll
    for (int j = 0; j < 16; j += 4) {
        float4 x = *(const float4*)(p + j);
        bf16_t h0 = (bf16_t)x.x, h1 = (bf16_t)x.y, h2 = (bf16_t)x.z, h3 = (bf16_t)x.w;
        hi[j+0] = h0; hi[j+1] = h1; hi[j+2] = h2; hi[j+3] = h3;
        lo[j+0] = (bf16_t)(x.x - (float)h0);
        lo[j+1] = (bf16_t)(x.y - (float)h1);
        lo[j+2] = (bf16_t)(x.z - (float)h2);
        lo[j+3] = (bf16_t)(x.w - (float)h3);
    }
}

// ============================ small kernels ================================

__global__ void k_zero_counters(int* counters) {
    if (threadIdx.x < 3) counters[threadIdx.x] = 0;
}

__global__ void k_partition(const int* __restrict__ idx, int n,
                            int* __restrict__ counters, int* __restrict__ lists) {
    int i = blockIdx.x * blockDim.x + threadIdx.x;
    if (i >= n) return;
    int id = idx[i];
    int c  = (id >= 40000) ? 2 : (id >= 20000) ? 1 : 0;
    int s  = atomicAdd(&counters[c], 1);
    lists[c * n + s] = i;   // list order nondeterministic; per-token output isn't
}

__global__ void k_split4(const float* __restrict__ src, bf16_t* __restrict__ hi,
                         bf16_t* __restrict__ lo, int n4) {
    int i = blockIdx.x * blockDim.x + threadIdx.x;
    if (i >= n4) return;
    float4 x = ((const float4*)src)[i];
    bfx4 h, l;
    h[0] = (bf16_t)x.x; h[1] = (bf16_t)x.y; h[2] = (bf16_t)x.z; h[3] = (bf16_t)x.w;
    l[0] = (bf16_t)(x.x - (float)h[0]);
    l[1] = (bf16_t)(x.y - (float)h[1]);
    l[2] = (bf16_t)(x.z - (float)h[2]);
    l[3] = (bf16_t)(x.w - (float)h[3]);
    ((bfx4*)hi)[i] = h;
    ((bfx4*)lo)[i] = l;
}

// ===================== partitioned bf16x3 GEMM kernel ======================
// Block: 32 same-cluster tokens x 256 output columns; 8 waves x (2 M x 2 N).
template <int C, bool PRE>
__launch_bounds__(256, 2)
__global__ void gemm_cluster(const int* __restrict__ idx,
                             const float* __restrict__ emb,
                             const float* __restrict__ w,
                             const bf16_t* __restrict__ whi,
                             const bf16_t* __restrict__ wlo,
                             const int* __restrict__ count_p,
                             const int* __restrict__ list,
                             float* __restrict__ out)
{
    constexpr int H   = cH[C];
    constexpr int LO  = cLO[C];
    constexpr int VSZ = cVS[C];
    constexpr int MT  = 32;   // tokens per block
    constexpr int KCH = 32;   // K per chunk
    constexpr int APD = 40;   // bf16 LDS row stride: 80B -> conflict-free
    constexpr int RPD = 36;   // raw f32 LDS row stride (TDM pads 4 dw/row)

    const int count = count_p[0];
    const int ty    = blockIdx.y;
    if (ty * MT >= count) return;             // uniform early-exit
    const int dblk  = blockIdx.x * 256;

    __shared__ int    s_slot[MT];
    __shared__ int    s_id[MT];               // local row ids for TDM gather
#if USE_TDM
    __shared__ float  s_raw[2 * MT * RPD];    // double-buffered raw f32 tiles
#endif
    __shared__ bf16_t s_hi[MT * APD];
    __shared__ bf16_t s_lo[MT * APD];

    const int tid = threadIdx.x;
    if (tid < MT) {
        int p  = ty * MT + tid;
        int sl = (p < count) ? list[p] : -1;
        s_slot[tid] = sl;
        s_id[tid]   = (sl >= 0) ? (idx[sl] - LO) : 0;
    }
    __syncthreads();

    // WMMA lane mapping (ISA 05_wmma 16-bit layouts)
    const int lane = tid & 31;
    const int l15  = lane & 15;
    const int half = lane >> 4;               // 0 / 1
    const int wv   = tid >> 5;                // wave 0..7
    const int dcol0 = dblk + (wv * 2 + 0) * 16 + l15;
    const int dcol1 = dblk + (wv * 2 + 1) * 16 + l15;

    // conversion/staging map: 256 threads cover 32 rows x 32 K (float4 each)
    const int srow = tid >> 3;                // 0..31
    const int skc  = (tid & 7) << 2;          // 0..28
    const int slot = s_slot[srow];

#if !USE_TDM
    const float* __restrict__ arow = nullptr;
    if (slot >= 0) arow = emb + (size_t)s_id[srow] * H + skc;
#endif

    // B panel bases: per lane 16 consecutive K at [k0 + 16*half)
    const size_t bo0 = (size_t)dcol0 * H + 16 * half;
    const size_t bo1 = (size_t)dcol1 * H + 16 * half;

#if USE_TDM
    // ---- TDM gather setup (scalar; wave 0 only issues) --------------------
    // Scalar wave test: readfirstlane -> SGPR compare -> s_cbranch (TDM
    // ignores EXEC, so an EXEC-masked branch would still issue the DMA).
    const int wvs = __builtin_amdgcn_readfirstlane((int)threadIdx.x) >> 5;
    i32x8 g1 = {};
    i32x4 g2[2] = {}, g3[2] = {};
    if (wvs == 0) {
        // D# group1: data_size=4B, pad_enable, pad every 32 dwords by 4 dwords
        g1[0] = (2 << 16) | (1 << 20) | (4 << 22) | (3 << 25);
        g1[1] = (H & 0xFFFF) << 16;                    // tensor_dim0 lo
        g1[2] = (H >> 16) | ((VSZ & 0xFFFF) << 16);    // dim0 hi | tensor_dim1 lo
        g1[3] = (VSZ >> 16) | (KCH << 16);             // dim1 hi | tile_dim0
        g1[4] = 16;                                    // tile_dim1 = #indices
        g1[5] = H;                                     // tensor_dim0_stride lo
        g1[6] = 0; g1[7] = 0;
        // D# groups 2/3: sixteen 16-bit row indices per row-group
#pragma unroll
        for (int g = 0; g < 2; ++g) {
            int pk[8];
#pragma unroll
            for (int j = 0; j < 8; ++j) {
                int a = s_id[g * 16 + 2 * j], b = s_id[g * 16 + 2 * j + 1];
                pk[j] = __builtin_amdgcn_readfirstlane((a & 0xFFFF) | (b << 16));
            }
            g2[g] = (i32x4){pk[0], pk[1], pk[2], pk[3]};
            g3[g] = (i32x4){pk[4], pk[5], pk[6], pk[7]};
        }
    }
    auto tdm_issue = [&](int k0, int par) {
        const unsigned long long ga =
            (unsigned long long)(uintptr_t)emb + (unsigned long long)k0 * 4u;
        const i32x8 gz = {};                           // extra operand (clang-23
                                                       // 6-arg form): zero fill
#pragma unroll
        for (int g = 0; g < 2; ++g) {
            unsigned lds = (unsigned)(uintptr_t)&s_raw[(par * 2 + g) * 16 * RPD];
            u32x4 g0;
            g0[0] = 1u | (1u << 31);                       // count=1, gather_mode
            g0[1] = lds;                                   // lds_addr
            g0[2] = (unsigned)ga;                          // global_addr[31:0]
            g0[3] = (unsigned)((ga >> 32) & 0x1FFFFFFu) | (2u << 30); // [56:32]|type
            __builtin_amdgcn_tensor_load_to_lds(g0, g1, g2[g], g3[g], gz, 0);
        }
    };
    if (wvs == 0) tdm_issue(0, 0);
    int par = 0;
#endif

    v8f acc[2][2] = {};

    for (int k0 = 0; k0 < H; k0 += KCH) {
        // ---- obtain raw A chunk ------------------------------------------
#if USE_TDM
        if (wvs == 0) __builtin_amdgcn_s_wait_tensorcnt(0);
        __syncthreads();                       // raw[par] ready; s_hi free
        if (wvs == 0 && (k0 + KCH) < H) tdm_issue(k0 + KCH, par ^ 1);
        float4 x = *(const float4*)&s_raw[(par * MT + srow) * RPD + skc];
        if (slot < 0) x = make_float4(0.f, 0.f, 0.f, 0.f);
        par ^= 1;
#else
        float4 x = make_float4(0.f, 0.f, 0.f, 0.f);
        if (slot >= 0) x = *(const float4*)(arow + k0);
#endif
        // ---- split and publish bf16 hi/lo --------------------------------
        bfx4 h4, l4;
        h4[0] = (bf16_t)x.x; h4[1] = (bf16_t)x.y;
        h4[2] = (bf16_t)x.z; h4[3] = (bf16_t)x.w;
        l4[0] = (bf16_t)(x.x - (float)h4[0]);
        l4[1] = (bf16_t)(x.y - (float)h4[1]);
        l4[2] = (bf16_t)(x.z - (float)h4[2]);
        l4[3] = (bf16_t)(x.w - (float)h4[3]);
        *(bfx4*)&s_hi[srow * APD + skc] = h4;
        *(bfx4*)&s_lo[srow * APD + skc] = l4;
        __syncthreads();                       // bf16 tiles ready

        // ---- A fragments: per lane K-group [16h,16h+16) -----
        v16bf ah[2], al[2];
#pragma unroll
        for (int m = 0; m < 2; ++m) {
            const int r = m * 16 + l15;
            ah[m] = cat8(*(const bfx8*)&s_hi[r * APD + 16 * half],
                         *(const bfx8*)&s_hi[r * APD + 16 * half + 8]);
            al[m] = cat8(*(const bfx8*)&s_lo[r * APD + 16 * half],
                         *(const bfx8*)&s_lo[r * APD + 16 * half + 8]);
        }

        // ---- B fragments --------------------------------------------------
        v16bf bh[2], bl[2];
        if constexpr (PRE) {
            bh[0] = ld16bf(whi + bo0 + k0);  bl[0] = ld16bf(wlo + bo0 + k0);
            bh[1] = ld16bf(whi + bo1 + k0);  bl[1] = ld16bf(wlo + bo1 + k0);
            if (k0 + KCH < H) {
                __builtin_prefetch(whi + bo0 + k0 + KCH, 0, 3);
                __builtin_prefetch(whi + bo1 + k0 + KCH, 0, 3);
            }
        } else {
            cvt16(w + bo0 + k0, bh[0], bl[0]);
            cvt16(w + bo1 + k0, bh[1], bl[1]);
        }

        // ---- 12 WMMAs: hi*hi + hi*lo + lo*hi, 4 independent acc between ---
#pragma unroll
        for (int m = 0; m < 2; ++m)
#pragma unroll
            for (int n = 0; n < 2; ++n)
                acc[m][n] = __builtin_amdgcn_wmma_f32_16x16x32_bf16(
                    false, ah[m], false, bh[n], (short)0, acc[m][n], false, false);
#pragma unroll
        for (int m = 0; m < 2; ++m)
#pragma unroll
            for (int n = 0; n < 2; ++n)
                acc[m][n] = __builtin_amdgcn_wmma_f32_16x16x32_bf16(
                    false, ah[m], false, bl[n], (short)0, acc[m][n], false, false);
#pragma unroll
        for (int m = 0; m < 2; ++m)
#pragma unroll
            for (int n = 0; n < 2; ++n)
                acc[m][n] = __builtin_amdgcn_wmma_f32_16x16x32_bf16(
                    false, al[m], false, bh[n], (short)0, acc[m][n], false, false);
#if !USE_TDM
        __syncthreads();                       // frag reads done; restage ok
#endif
    }

    // ---- scatter rows: C/D layout (M = v + 8*half, N = l15) ---------------
#pragma unroll
    for (int m = 0; m < 2; ++m) {
#pragma unroll
        for (int v = 0; v < 8; ++v) {
            const int r = m * 16 + half * 8 + v;
            const int t = s_slot[r];
            if (t >= 0) {
                __builtin_nontemporal_store(acc[m][0][v], out + (size_t)t * D_MODEL + dcol0);
                __builtin_nontemporal_store(acc[m][1][v], out + (size_t)t * D_MODEL + dcol1);
            }
        }
    }
}

// =============== fallback: padded f32 16x16x4 WMMA (no workspace) ==========

__launch_bounds__(256, 2)
__global__ void adaptive_emb_f32(const int* __restrict__ idx,
                                 const float* __restrict__ emb0, const float* __restrict__ w0,
                                 const float* __restrict__ emb1, const float* __restrict__ w1,
                                 const float* __restrict__ emb2, const float* __restrict__ w2,
                                 float* __restrict__ out, int n_tok)
{
    constexpr int PADF = 68;
    __shared__ int   s_ids[16];
    __shared__ float s_A[16 * PADF];

    const int tid  = threadIdx.x;
    const int tok0 = blockIdx.y * 16;
    const int dblk = blockIdx.x * 256;

    if (tid < 16) {
        int g = tok0 + tid;
        s_ids[tid] = (g < n_tok) ? idx[g] : 0;
    }
    __syncthreads();

    const int lane = tid & 31;
    const int l15  = lane & 15;
    const int lh2  = (lane >> 4) << 1;
    const int wv   = tid >> 5;
    const int dcol0 = dblk + (wv * 2 + 0) * 16 + l15;
    const int dcol1 = dblk + (wv * 2 + 1) * 16 + l15;
    const int srow = tid >> 4;
    const int scol = (tid & 15) << 2;

    v8f acc0 = {}, acc1 = {};
    const float* embs[3] = {emb0, emb1, emb2};
    const float* ws[3]   = {w0, w1, w2};

#pragma unroll
    for (int c = 0; c < 3; ++c) {
        const int lo = cLO[c], hic = cHIe[c], h = cH[c];
        bool any = false;
        for (int r = 0; r < 16; ++r) {
            int id = s_ids[r];
            any = any || (id >= lo && id < hic);
        }
        if (!any) continue;

        const int  myid   = s_ids[srow];
        const bool member = (myid >= lo) && (myid < hic);
        const float* __restrict__ arow = embs[c] + (member ? (size_t)(myid - lo) * h : 0) + scol;
        const float* __restrict__ b0   = ws[c] + (size_t)dcol0 * h + lh2;
        const float* __restrict__ b1   = ws[c] + (size_t)dcol1 * h + lh2;

        for (int k0 = 0; k0 < h; k0 += 64) {
            float4 av = make_float4(0.f, 0.f, 0.f, 0.f);
            if (member) av = *(const float4*)(arow + k0);
            *(float4*)&s_A[srow * PADF + scol] = av;
            __syncthreads();
#pragma unroll
            for (int kk = 0; kk < 64; kk += 4) {
                v2f a   = *(const v2f*)&s_A[l15 * PADF + kk + lh2];
                v2f vb0 = *(const v2f*)(b0 + k0 + kk);
                v2f vb1 = *(const v2f*)(b1 + k0 + kk);
                acc0 = __builtin_amdgcn_wmma_f32_16x16x4_f32(false, a, false, vb0, (short)0, acc0, false, false);
                acc1 = __builtin_amdgcn_wmma_f32_16x16x4_f32(false, a, false, vb1, (short)0, acc1, false, false);
            }
            __syncthreads();
        }
    }

    const int mbase = tok0 + ((lane >> 4) << 3);
#pragma unroll
    for (int v = 0; v < 8; ++v) {
        int m = mbase + v;
        if (m < n_tok) {
            __builtin_nontemporal_store(acc0[v], out + (size_t)m * D_MODEL + dcol0);
            __builtin_nontemporal_store(acc1[v], out + (size_t)m * D_MODEL + dcol1);
        }
    }
}

// ================================ host =====================================

extern "C" void kernel_launch(void* const* d_in, const int* in_sizes, int n_in,
                              void* d_out, int out_size, void* d_ws, size_t ws_size,
                              hipStream_t stream) {
    const int*   idx  = (const int*)d_in[0];
    const float* emb0 = (const float*)d_in[1];
    const float* w0   = (const float*)d_in[2];
    const float* emb1 = (const float*)d_in[3];
    const float* w1   = (const float*)d_in[4];
    const float* emb2 = (const float*)d_in[5];
    const float* w2   = (const float*)d_in[6];
    float* out = (float*)d_out;
    const int n = in_sizes[0];                       // 16384 tokens

    // workspace: [counters(3) pad 256B][lists 3*n ints][w hi/lo bf16]
    int* counters = (int*)d_ws;
    int* lists    = (int*)((char*)d_ws + 256);
    const size_t wElems[3] = {(size_t)D_MODEL * 1024, (size_t)D_MODEL * 256,
                              (size_t)D_MODEL * 64};
    size_t off_w = (256 + (size_t)3 * n * sizeof(int) + 255) & ~(size_t)255;
    bf16_t *whi[3], *wlo[3];
    {
        char* p = (char*)d_ws + off_w;
        for (int c = 0; c < 3; ++c) {
            whi[c] = (bf16_t*)p; p += wElems[c] * sizeof(bf16_t);
            wlo[c] = (bf16_t*)p; p += wElems[c] * sizeof(bf16_t);
        }
        size_t need_pre  = (size_t)(p - (char*)d_ws);
        size_t need_part = off_w;

        if (ws_size >= need_part) {
            k_zero_counters<<<1, 64, 0, stream>>>(counters);
            k_partition<<<(n + 255) / 256, 256, 0, stream>>>(idx, n, counters, lists);

            const bool pre = (ws_size >= need_pre);
            if (pre) {
                const float* wsrc[3] = {w0, w1, w2};
                for (int c = 0; c < 3; ++c) {
                    int n4 = (int)(wElems[c] / 4);
                    k_split4<<<(n4 + 255) / 256, 256, 0, stream>>>(wsrc[c], whi[c], wlo[c], n4);
                }
            }
            dim3 grid(D_MODEL / 256, (n + 31) / 32);
            if (pre) {
                gemm_cluster<0, true><<<grid, 256, 0, stream>>>(idx, emb0, w0, whi[0], wlo[0], counters + 0, lists + 0 * (size_t)n, out);
                gemm_cluster<1, true><<<grid, 256, 0, stream>>>(idx, emb1, w1, whi[1], wlo[1], counters + 1, lists + 1 * (size_t)n, out);
                gemm_cluster<2, true><<<grid, 256, 0, stream>>>(idx, emb2, w2, whi[2], wlo[2], counters + 2, lists + 2 * (size_t)n, out);
            } else {
                gemm_cluster<0, false><<<grid, 256, 0, stream>>>(idx, emb0, w0, nullptr, nullptr, counters + 0, lists + 0 * (size_t)n, out);
                gemm_cluster<1, false><<<grid, 256, 0, stream>>>(idx, emb1, w1, nullptr, nullptr, counters + 1, lists + 1 * (size_t)n, out);
                gemm_cluster<2, false><<<grid, 256, 0, stream>>>(idx, emb2, w2, nullptr, nullptr, counters + 2, lists + 2 * (size_t)n, out);
            }
            return;
        }
    }
    // tiny-workspace fallback: padded f32 WMMA
    dim3 grid(D_MODEL / 256, (n + 15) / 16);
    adaptive_emb_f32<<<grid, 256, 0, stream>>>(idx, emb0, w0, emb1, w1, emb2, w2, out, n);
}
